// UnquantizedSparseMoELayer_8177617732174
// MI455X (gfx1250) — matrix operand order repaired and checked by reading, not compiled, not run on hardware
//
#include <hip/hip_runtime.h>
#include <hip/hip_bf16.h>
#include <cstdint>
#include <cstddef>

// Problem constants (match reference)
#define T_TOK  4096
#define DMODEL 2048
#define DFF    8192
#define NEXP   8

#define KCH        128   // K elements staged per barrier
#define LDS_STRIDE 136   // bf16 elements per LDS row (272B = 256B data + 16B pad)

typedef __bf16 v16bf __attribute__((ext_vector_type(16)));
typedef __bf16 v8bf  __attribute__((ext_vector_type(8)));
typedef float  v8f   __attribute__((ext_vector_type(8)));
typedef float  v4f   __attribute__((ext_vector_type(4)));
typedef unsigned int u32x4 __attribute__((ext_vector_type(4)));
typedef int    i32x8 __attribute__((ext_vector_type(8)));
typedef int    i32x4 __attribute__((ext_vector_type(4)));
typedef int    vi4   __attribute__((vector_size(16)));   // matches builtin param type

union FragBF { v16bf v; v8bf h[2]; };

#define AS_GLOBAL __attribute__((address_space(1)))
#define AS_LDS    __attribute__((address_space(3)))

#if defined(__AMDGCN__) && __has_builtin(__builtin_amdgcn_global_load_async_to_lds_b128)
#define HAVE_ASYNC_LDS 1
#else
#define HAVE_ASYNC_LDS 0
#endif

#if defined(__AMDGCN__) && __has_builtin(__builtin_amdgcn_tensor_load_to_lds)
#define HAVE_TDM 1
#else
#define HAVE_TDM 0
#endif

// ---- workspace layout (bytes) ----
static constexpr size_t WS_COUNTS  = 0;                                   // 8 * 4
static constexpr size_t WS_OFFSETS = 256;                                 // 9 * 4
static constexpr size_t WS_LISTS   = 4096;                                // 8*4096*4
static constexpr size_t WS_COEFS   = WS_LISTS + (size_t)NEXP * T_TOK * 4; // 8*4096*4
static constexpr size_t WS_XBF     = WS_COEFS + (size_t)NEXP * T_TOK * 4; // 4096*2048*2
static constexpr size_t WS_H       = WS_XBF + (size_t)T_TOK * DMODEL * 2; // 8192*8192*2

// ---------------- fragment helpers ----------------

// A fragment from LDS tile (16-bit A 16x32, contiguous k order matching B):
// lanes 0-15: row M=lane, K kk..kk+15
// lanes 16-31: same rows, K kk+16..kk+31
__device__ inline v16bf load_a_lds(const __bf16* tile, int m, int kk, int lane) {
    const __bf16* rowptr = tile + (size_t)(m * 16 + (lane & 15)) * LDS_STRIDE;
    int kofs = kk + ((lane >> 4) << 4);
    FragBF f;
    f.h[0] = *reinterpret_cast<const v8bf*>(rowptr + kofs);
    f.h[1] = *reinterpret_cast<const v8bf*>(rowptr + kofs + 8);
    return f.v;
}

// B fragment, 16-bit B 32x16, contiguous k order matching A: lane n (0-15) holds column N=n, K=k0..k0+15
// (16 consecutive); lanes 16-31 hold K=k0+16..k0+31. Weights are [N,K] row-major
// fp32: 64B contiguous per lane, converted to bf16 in-register (v_cvt_pk_bf16_f32).
__device__ inline v16bf load_b_f32(const float* base, long ld, int k0, int lane) {
    int r = lane & 15;
    int k = k0 + ((lane >> 4) << 4);
    const float* p = base + (long)r * ld + k;
    const v4f* pv = reinterpret_cast<const v4f*>(p);
    v4f c0 = pv[0], c1 = pv[1], c2 = pv[2], c3 = pv[3];
    v16bf f;
#pragma unroll
    for (int i = 0; i < 4; ++i) {
        f[i]      = (__bf16)c0[i];
        f[4 + i]  = (__bf16)c1[i];
        f[8 + i]  = (__bf16)c2[i];
        f[12 + i] = (__bf16)c3[i];
    }
    return f;
}

// Cooperative LDS fill, one thread owns half a row (64 bf16 = 128B = 8 x b128).
__device__ inline void fill_lds_half_row(__bf16* dst, const __bf16* src) {
#if HAVE_ASYNC_LDS
    const AS_GLOBAL __bf16* g = (const AS_GLOBAL __bf16*)src;
    AS_LDS __bf16* l = (AS_LDS __bf16*)dst;
#pragma unroll
    for (int c = 0; c < 8; ++c) {
        __builtin_amdgcn_global_load_async_to_lds_b128(
            (AS_GLOBAL vi4*)(g + c * 8), (AS_LDS vi4*)(l + c * 8), 0, 0);
    }
    asm volatile("s_wait_asynccnt 0x0" ::: "memory");
#else
#pragma unroll
    for (int c = 0; c < 8; ++c) {
        *reinterpret_cast<v8bf*>(dst + c * 8) =
            *reinterpret_cast<const v8bf*>(src + c * 8);
    }
#endif
}

// ---------------- small kernels ----------------

__global__ void moe_init(float* __restrict__ out, int* __restrict__ counts) {
    size_t id = (size_t)blockIdx.x * blockDim.x + threadIdx.x;
    if (id < (size_t)T_TOK * DMODEL) out[id] = 0.0f;
    if (id < NEXP) counts[id] = 0;
}

__global__ void moe_cvt_x(const float* __restrict__ x, __bf16* __restrict__ xbf) {
    size_t id = (size_t)blockIdx.x * blockDim.x + threadIdx.x;
    if (id < (size_t)T_TOK * DMODEL) xbf[id] = (__bf16)x[id];
}

__global__ void moe_router(const float* __restrict__ gating,
                           float* __restrict__ coefs, int* __restrict__ lists,
                           int* __restrict__ counts) {
    int t = blockIdx.x * blockDim.x + threadIdx.x;
    if (t >= T_TOK) return;
    float g[NEXP];
    float m = -3.4e38f;
#pragma unroll
    for (int i = 0; i < NEXP; ++i) { g[i] = gating[t * NEXP + i]; m = fmaxf(m, g[i]); }
    float p[NEXP];
    float s = 0.0f;
#pragma unroll
    for (int i = 0; i < NEXP; ++i) { p[i] = __expf(g[i] - m); s += p[i]; }
    int i1 = 0;
#pragma unroll
    for (int i = 1; i < NEXP; ++i) if (p[i] > p[i1]) i1 = i;
    int i2 = (i1 == 0) ? 1 : 0;
#pragma unroll
    for (int i = 0; i < NEXP; ++i) if (i != i1 && p[i] > p[i2]) i2 = i;
    float w1 = p[i1] / s, w2 = p[i2] / s;
    float rs = w1 + w2;
    w1 /= rs; w2 /= rs;
    int pos1 = atomicAdd(&counts[i1], 1);
    lists[i1 * T_TOK + pos1] = t;
    coefs[i1 * T_TOK + pos1] = w1;
    int pos2 = atomicAdd(&counts[i2], 1);
    lists[i2 * T_TOK + pos2] = t;
    coefs[i2 * T_TOK + pos2] = w2;
}

__global__ void moe_prefix(const int* __restrict__ counts, int* __restrict__ offs) {
    if (threadIdx.x == 0 && blockIdx.x == 0) {
        int s = 0;
#pragma unroll
        for (int e = 0; e < NEXP; ++e) { offs[e] = s; s += counts[e]; }
        offs[NEXP] = s;
    }
}

// ---------------- GEMM1: h = silu(x Wg^T) * (x Wu^T), gathered rows ----------------
// block = 128 threads (4 waves). Tile: 64 rows x 64 cols (gate + up).
// Wave w owns N-strip [j0 + 16w, +16): weights loaded exactly once per block.
// A tile (gathered bf16 token rows) staged in LDS via async-to-LDS.
__global__ __launch_bounds__(128) void moe_gemm1(
    const float* __restrict__ gup, const __bf16* __restrict__ xbf,
    const int* __restrict__ counts, const int* __restrict__ offs,
    const int* __restrict__ lists, __bf16* __restrict__ hbuf) {
    __shared__ __bf16 lds_a[64 * LDS_STRIDE];

    int e = blockIdx.z;
    int cnt = counts[e];
    int mbase = blockIdx.x * 64;
    if (mbase >= cnt) return;
    int j0 = blockIdx.y * 64;
    int wave = threadIdx.x >> 5;
    int lane = threadIdx.x & 31;

    // fill role: thread owns half of one LDS row (gather source via token list)
    int fr  = threadIdx.x >> 1;
    int hlf = threadIdx.x & 1;
    int frc = mbase + fr;
    frc = frc < cnt ? frc : cnt - 1;
    int ftok = lists[e * T_TOK + frc];
    const __bf16* fsrc = xbf + (size_t)ftok * DMODEL + hlf * 64;
    __bf16* fdst = lds_a + (size_t)fr * LDS_STRIDE + hlf * 64;

    const float* wbase = gup + (size_t)e * (2 * (size_t)DFF) * DMODEL;
    const float* wgate = wbase + (size_t)(j0 + wave * 16) * DMODEL;
    const float* wup   = wbase + ((size_t)DFF + j0 + wave * 16) * DMODEL;

    v8f zero = {0.f, 0.f, 0.f, 0.f, 0.f, 0.f, 0.f, 0.f};
    v8f accg[4] = {zero, zero, zero, zero};
    v8f accu[4] = {zero, zero, zero, zero};

    for (int k0 = 0; k0 < DMODEL; k0 += KCH) {
        if (k0) __syncthreads();               // readers done with previous tile
        fill_lds_half_row(fdst, fsrc + k0);
        __syncthreads();                       // tile visible to all waves
        __builtin_prefetch(wgate + (size_t)(lane & 15) * DMODEL + k0 + KCH, 0, 0);
#pragma unroll
        for (int kk = 0; kk < KCH; kk += 32) {
            v16bf a0 = load_a_lds(lds_a, 0, kk, lane);
            v16bf a1 = load_a_lds(lds_a, 1, kk, lane);
            v16bf a2 = load_a_lds(lds_a, 2, kk, lane);
            v16bf a3 = load_a_lds(lds_a, 3, kk, lane);
            v16bf bg = load_b_f32(wgate, DMODEL, k0 + kk, lane);
            v16bf bu = load_b_f32(wup,   DMODEL, k0 + kk, lane);
            accg[0] = __builtin_amdgcn_wmma_f32_16x16x32_bf16(false, a0, false, bg, (short)0, accg[0], false, false);
            accu[0] = __builtin_amdgcn_wmma_f32_16x16x32_bf16(false, a0, false, bu, (short)0, accu[0], false, false);
            accg[1] = __builtin_amdgcn_wmma_f32_16x16x32_bf16(false, a1, false, bg, (short)0, accg[1], false, false);
            accu[1] = __builtin_amdgcn_wmma_f32_16x16x32_bf16(false, a1, false, bu, (short)0, accu[1], false, false);
            accg[2] = __builtin_amdgcn_wmma_f32_16x16x32_bf16(false, a2, false, bg, (short)0, accg[2], false, false);
            accu[2] = __builtin_amdgcn_wmma_f32_16x16x32_bf16(false, a2, false, bu, (short)0, accu[2], false, false);
            accg[3] = __builtin_amdgcn_wmma_f32_16x16x32_bf16(false, a3, false, bg, (short)0, accg[3], false, false);
            accu[3] = __builtin_amdgcn_wmma_f32_16x16x32_bf16(false, a3, false, bu, (short)0, accu[3], false, false);
        }
    }

    int hbase = offs[e];
    int col = j0 + wave * 16 + (lane & 15);
#pragma unroll
    for (int m = 0; m < 4; ++m) {
#pragma unroll
        for (int v = 0; v < 8; ++v) {
            int r = mbase + m * 16 + v + ((lane >> 4) << 3);
            if (r < cnt) {
                float g = accg[m][v];
                float u = accu[m][v];
                float hv = (g / (1.0f + __expf(-g))) * u;  // silu(g) * u
                hbuf[(size_t)(hbase + r) * DFF + col] = (__bf16)hv;
            }
        }
    }
}

// ---------------- GEMM2: out[tok,:] += coef * (h Wd^T) ----------------
// A tile (contiguous bf16 h rows) staged in LDS via the Tensor Data Mover.
__global__ __launch_bounds__(128) void moe_gemm2(
    const float* __restrict__ wdown, const __bf16* __restrict__ hbuf,
    const int* __restrict__ counts, const int* __restrict__ offs,
    const int* __restrict__ lists, const float* __restrict__ coefs,
    float* __restrict__ out) {
    __shared__ __bf16 lds_a[64 * LDS_STRIDE];

    int e = blockIdx.z;
    int cnt = counts[e];
    int mbase = blockIdx.x * 64;
    if (mbase >= cnt) return;
    int n0 = blockIdx.y * 64;
    int wave = threadIdx.x >> 5;
    int lane = threadIdx.x & 31;
    int hbase = offs[e];
    int off8 = offs[NEXP];

    const float* wd = wdown + (size_t)e * DMODEL * DFF + (size_t)(n0 + wave * 16) * DFF;

#if !HAVE_TDM
    int fr  = threadIdx.x >> 1;
    int hlf = threadIdx.x & 1;
    int frc = mbase + fr;
    frc = frc < cnt ? frc : cnt - 1;
    const __bf16* fsrc = hbuf + (size_t)(hbase + frc) * DFF + hlf * 64;
    __bf16* fdst = lds_a + (size_t)fr * LDS_STRIDE + hlf * 64;
#endif

    v8f zero = {0.f, 0.f, 0.f, 0.f, 0.f, 0.f, 0.f, 0.f};
    v8f acc[4] = {zero, zero, zero, zero};

    for (int k0 = 0; k0 < DFF; k0 += KCH) {
        if (k0) __syncthreads();
#if HAVE_TDM
        if (wave == 0) {
            // Tensor DMA descriptor: 2D tile 128(K) x rows, bf16, padded LDS rows
            int rows = off8 - (hbase + mbase);
            if (rows > 64) rows = 64;
            unsigned ldsoff = (unsigned)(unsigned long long)(AS_LDS __bf16*)lds_a;
            unsigned long long ga = (unsigned long long)(uintptr_t)(
                hbuf + (size_t)(hbase + mbase) * DFF + k0);
            u32x4 g0;
            g0.x = 1u;                                    // count=1, normal mode
            g0.y = ldsoff;                                // lds_addr
            g0.z = (unsigned)(ga & 0xffffffffull);        // global_addr[31:0]
            g0.w = (unsigned)((ga >> 32) & 0x1ffffffull)  // global_addr[56:32]
                 | (2u << 30);                            // type=2 (image)
            i32x8 g1;
            g1[0] = (1 << 16)      // data_size = 2 bytes
                  | (1 << 20)      // pad_enable
                  | (5 << 22)      // pad_interval: 64 DWORDs (256B)
                  | (3 << 25);     // pad_amount: 4 DWORDs (16B)
            g1[1] = (int)(128u << 16);                // tensor_dim0 = 128 (lo16)
            g1[2] = (int)((unsigned)rows << 16);      // tensor_dim1 = rows (lo16)
            g1[3] = (int)(128u << 16);                // tile_dim0 = 128
            g1[4] = rows;                             // tile_dim1 = rows, tile_dim2 = 0
            g1[5] = DFF;                              // tensor_dim0_stride (elements)
            g1[6] = 0;
            g1[7] = 0;
            i32x4 gz = {0, 0, 0, 0};
#if __clang_major__ >= 23
            i32x8 gz8 = {0, 0, 0, 0, 0, 0, 0, 0};
            __builtin_amdgcn_tensor_load_to_lds(g0, g1, gz, gz, gz8, 0);
#else
            __builtin_amdgcn_tensor_load_to_lds(g0, g1, gz, gz, 0);
#endif
            asm volatile("s_wait_tensorcnt 0x0" ::: "memory");
        }
        __syncthreads();
#else
        fill_lds_half_row(fdst, fsrc + k0);
        __syncthreads();
#endif
        __builtin_prefetch(wd + (size_t)(lane & 15) * DFF + k0 + KCH, 0, 0);
#pragma unroll
        for (int kk = 0; kk < KCH; kk += 32) {
            v16bf a0 = load_a_lds(lds_a, 0, kk, lane);
            v16bf a1 = load_a_lds(lds_a, 1, kk, lane);
            v16bf a2 = load_a_lds(lds_a, 2, kk, lane);
            v16bf a3 = load_a_lds(lds_a, 3, kk, lane);
            v16bf b = load_b_f32(wd, DFF, k0 + kk, lane);
            acc[0] = __builtin_amdgcn_wmma_f32_16x16x32_bf16(false, a0, false, b, (short)0, acc[0], false, false);
            acc[1] = __builtin_amdgcn_wmma_f32_16x16x32_bf16(false, a1, false, b, (short)0, acc[1], false, false);
            acc[2] = __builtin_amdgcn_wmma_f32_16x16x32_bf16(false, a2, false, b, (short)0, acc[2], false, false);
            acc[3] = __builtin_amdgcn_wmma_f32_16x16x32_bf16(false, a3, false, b, (short)0, acc[3], false, false);
        }
    }

    int col = n0 + wave * 16 + (lane & 15);
#pragma unroll
    for (int m = 0; m < 4; ++m) {
#pragma unroll
        for (int v = 0; v < 8; ++v) {
            int r = mbase + m * 16 + v + ((lane >> 4) << 3);
            if (r < cnt) {
                int tok = lists[e * T_TOK + r];
                float c = coefs[e * T_TOK + r];
                atomicAdd(out + (size_t)tok * DMODEL + col, c * acc[m][v]);
            }
        }
    }
}

// ---------------- launch ----------------
extern "C" void kernel_launch(void* const* d_in, const int* in_sizes, int n_in,
                              void* d_out, int out_size, void* d_ws, size_t ws_size,
                              hipStream_t stream) {
    const float* x      = (const float*)d_in[0];
    const float* gating = (const float*)d_in[1];
    const float* gup    = (const float*)d_in[2];
    const float* wdown  = (const float*)d_in[3];
    // d_in[4] = topk scalar (fixed at 2 here)

    char* ws = (char*)d_ws;
    int*    counts = (int*)(ws + WS_COUNTS);
    int*    offs   = (int*)(ws + WS_OFFSETS);
    int*    lists  = (int*)(ws + WS_LISTS);
    float*  coefs  = (float*)(ws + WS_COEFS);
    __bf16* xbf    = (__bf16*)(ws + WS_XBF);
    __bf16* hbuf   = (__bf16*)(ws + WS_H);
    float*  out    = (float*)d_out;

    const int n_elem_blocks = (T_TOK * DMODEL + 255) / 256;
    moe_init<<<n_elem_blocks, 256, 0, stream>>>(out, counts);
    moe_cvt_x<<<n_elem_blocks, 256, 0, stream>>>(x, xbf);
    moe_router<<<(T_TOK + 255) / 256, 256, 0, stream>>>(gating, coefs, lists, counts);
    moe_prefix<<<1, 32, 0, stream>>>(counts, offs);

    dim3 g1(T_TOK / 64, DFF / 64, NEXP);
    moe_gemm1<<<g1, 128, 0, stream>>>(gup, xbf, counts, offs, lists, hbuf);

    dim3 g2(T_TOK / 64, DMODEL / 64, NEXP);
    moe_gemm2<<<g2, 128, 0, stream>>>(wdown, hbuf, counts, offs, lists, coefs, out);
}
